// BatchHetInfLinkPred_59691455479895
// MI455X (gfx1250) — matrix operand order repaired and checked by reading, not compiled, not run on hardware
//
#include <hip/hip_runtime.h>
#include <hip/hip_bf16.h>

// ---------------------------------------------------------------------------
// HAN/GAT forward for MI455X (gfx1250, wave32).
// Dense projections: LDS-staged f16 WMMA GEMM (v_wmma_f32_16x16x32_f16),
// 8 waves share a 128x128 tile staged in 132KB of the WGP's 320KB LDS.
// Edge softmax/scatter via coalesced atomic passes. Graph-capture safe.
// ---------------------------------------------------------------------------

typedef _Float16 v8h  __attribute__((ext_vector_type(8)));
typedef _Float16 v16h __attribute__((ext_vector_type(16)));
typedef float    v8f  __attribute__((ext_vector_type(8)));

union H16 { v16h v; v8h h[2]; };

#define NEG_SLOPE 0.2f
#define EPS_F 1e-16f

#define KMAX 256
#define KPAD 8
#define LDWMAX (KMAX + KPAD)   // 264 halfs; 132 DWORDs == 4 mod 64 banks

// ---- helpers ---------------------------------------------------------------

__device__ __forceinline__ unsigned fenc(float f) {
  unsigned u = __float_as_uint(f);
  return (u & 0x80000000u) ? ~u : (u | 0x80000000u);
}
__device__ __forceinline__ float fdec(unsigned u) {
  return __uint_as_float((u & 0x80000000u) ? (u & 0x7FFFFFFFu) : ~u);
}
__device__ __forceinline__ float lrelu(float v) { return v > 0.f ? v : NEG_SLOPE * v; }

// ---- elementwise conversions ----------------------------------------------

__global__ void k_f32_to_f16(const float* __restrict__ in, _Float16* __restrict__ out,
                             long long n) {
  long long i = (long long)blockIdx.x * blockDim.x + threadIdx.x;
  if (i < n) out[i] = (_Float16)in[i];
}

// Wt[n][k] = W[k][n]  (f32 row-major K x Nc  ->  f16 row-major Nc x K)
__global__ void k_conv_w_t(const float* __restrict__ W, _Float16* __restrict__ Wt,
                           int K, int Nc) {
  int i = blockIdx.x * blockDim.x + threadIdx.x;
  if (i < K * Nc) {
    int k = i / Nc, n = i - k * Nc;
    Wt[(long long)n * K + k] = (_Float16)W[i];
  }
}

// ---- WMMA GEMM: out = act(A[M,K] @ W[K,Nc] + bias) -------------------------
// Block = 256 threads (8 waves) computes a 128-row x 128-col tile.
// A tile (128xK) and B panel (128 cols x K, from transposed Wt) are staged in
// LDS once per block; each wave computes one 16-row strip against all 8
// column tiles from LDS (ds_load_b128, bank-conflict-free via +8-half pad).
// outF (f32) and/or outH (f16) optional. act: 0=none, 1=relu.

__global__ void __launch_bounds__(256)
k_wmma_gemm(const _Float16* __restrict__ A, const _Float16* __restrict__ Wt,
            const float* __restrict__ bias, float* __restrict__ outF,
            _Float16* __restrict__ outH, int M, int K, int Nc, int act) {
  __shared__ _Float16 sA[128 * LDWMAX];
  __shared__ _Float16 sB[128 * LDWMAX];

  const int tid = threadIdx.x;
  const int rowblk = blockIdx.x * 128;
  const int colblk = blockIdx.y * 128;
  const int ldw = K + KPAD;

  // ---- stage A tile (128 x K) and B panel (128 cols x K) into LDS ----
  for (int l = tid * 8; l < 128 * K; l += 256 * 8) {
    int r = l / K, kk = l - r * K;
    int gr = rowblk + r;
    if (gr >= M) gr = M - 1;                       // clamp: keeps EXEC uniform
    *(v8h*)&sA[r * ldw + kk] = *(const v8h*)(A + (long long)gr * K + kk);
  }
  for (int l = tid * 8; l < 128 * K; l += 256 * 8) {
    int c = l / K, kk = l - c * K;
    *(v8h*)&sB[c * ldw + kk] = *(const v8h*)(Wt + (long long)(colblk + c) * K + kk);
  }
  __syncthreads();

  // ---- per-wave 16x128 strip ----
  const int w = tid >> 5;                // wave id 0..7 -> row strip
  const int ln = tid & 31;
  const int hi = ln >> 4, lo = ln & 15;

  // A fragment (16x32 f16): lanes 0-15 rows, K chunks {0-7,16-23};
  // lanes 16-31 same rows, K chunks {8-15,24-31}.
  const _Float16* aptr = &sA[(w * 16 + lo) * ldw + hi * 8];
  // B fragment (32x16 f16): lane = col, lanes 0-15 K chunks {0-7,16-23}, 16-31 K chunks {8-15,24-31}
  const _Float16* bptr = &sB[lo * ldw + hi * 8];

  v8f acc[8] = {};

  for (int k = 0; k < K; k += 32) {
    H16 a;
    a.h[0] = *(const v8h*)(aptr + k);
    a.h[1] = *(const v8h*)(aptr + k + 16);
#pragma unroll
    for (int t = 0; t < 8; ++t) {
      const _Float16* bp = bptr + t * 16 * ldw + k;
      H16 b;
      b.h[0] = *(const v8h*)bp;
      b.h[1] = *(const v8h*)(bp + 16);
      acc[t] = __builtin_amdgcn_wmma_f32_16x16x32_f16(
          false, a.v, false, b.v, (short)0, acc[t], false, false);
    }
  }

  // ---- epilogue: bias + activation, guarded rows ----
#pragma unroll
  for (int t = 0; t < 8; ++t) {
    int gcol = colblk + t * 16 + lo;
    float bv = bias ? bias[gcol] : 0.f;
#pragma unroll
    for (int r = 0; r < 8; ++r) {
      int grow = rowblk + w * 16 + hi * 8 + r;  // C/D layout: VGPR r -> row r+8*hi
      if (grow < M) {
        float v = acc[t][r] + bv;
        if (act) v = v > 0.f ? v : 0.f;
        long long o = (long long)grow * Nc + gcol;
        if (outF) outF[o] = v;
        if (outH) outH[o] = (_Float16)v;
      }
    }
  }
}

// ---- per-(node,head) attention coefficients --------------------------------
// One wave per (node, head): AS = <h, a_src>, AD = <h, a_dst>.

__global__ void k_alpha(const float* __restrict__ Hm, const float* __restrict__ a_s,
                        const float* __restrict__ a_d, float* __restrict__ AS,
                        float* __restrict__ AD, int Nn, int Hh, int C) {
  int wid = (int)((blockIdx.x * blockDim.x + threadIdx.x) >> 5);
  int lane = threadIdx.x & 31;
  if (wid >= Nn * Hh) return;
  int n = wid / Hh, h = wid - n * Hh;
  const float* row = Hm + (long long)n * Hh * C + (long long)h * C;
  float s1 = 0.f, s2 = 0.f;
  for (int c = lane; c < C; c += 32) {
    float v = row[c];
    s1 += v * a_s[h * C + c];
    s2 += v * a_d[h * C + c];
  }
#pragma unroll
  for (int off = 16; off > 0; off >>= 1) {
    s1 += __shfl_xor(s1, off, 32);
    s2 += __shfl_xor(s2, off, 32);
  }
  if (lane == 0) { AS[wid] = s1; AD[wid] = s2; }
}

// ---- edge passes -----------------------------------------------------------
// Edge e in [0,E): (src[e], dst[e]); e in [E, E+N): self loop (e-E, e-E).

__global__ void k_edge_max(const int* __restrict__ src, const int* __restrict__ dst,
                           int E, int Ntot, const float* __restrict__ AS,
                           const float* __restrict__ AD, unsigned* __restrict__ Mx,
                           int Hh) {
  long long idx = (long long)blockIdx.x * blockDim.x + threadIdx.x;
  if (idx >= (long long)Ntot * Hh) return;
  int e = (int)(idx / Hh), h = (int)(idx - (long long)e * Hh);
  int s, d;
  if (e < E) { s = src[e]; d = dst[e]; } else { s = d = e - E; }
  float v = lrelu(AS[s * Hh + h] + AD[d * Hh + h]);
  atomicMax(&Mx[d * Hh + h], fenc(v));
}

__global__ void k_edge_exp(const int* __restrict__ src, const int* __restrict__ dst,
                           int E, int Ntot, const float* __restrict__ AS,
                           const float* __restrict__ AD, const unsigned* __restrict__ Mx,
                           float* __restrict__ DEN, float* __restrict__ EX, int Hh) {
  long long idx = (long long)blockIdx.x * blockDim.x + threadIdx.x;
  if (idx >= (long long)Ntot * Hh) return;
  int e = (int)(idx / Hh), h = (int)(idx - (long long)e * Hh);
  int s, d;
  if (e < E) { s = src[e]; d = dst[e]; } else { s = d = e - E; }
  float v = lrelu(AS[s * Hh + h] + AD[d * Hh + h]);
  float m = fdec(Mx[d * Hh + h]);
  float ex = __expf(v - m);
  EX[idx] = ex;
  atomicAdd(&DEN[d * Hh + h], ex);
}

__global__ void k_edge_scatter(const int* __restrict__ src, const int* __restrict__ dst,
                               int E, int Ntot, const float* __restrict__ Hm,
                               const float* __restrict__ EX, const float* __restrict__ DEN,
                               float* __restrict__ ACC, int Hh, int C) {
  long long idx = (long long)blockIdx.x * blockDim.x + threadIdx.x;
  long long tot = (long long)Ntot * Hh * C;
  if (idx >= tot) return;
  int c = (int)(idx % C);
  long long eh = idx / C;
  int h = (int)(eh % Hh);
  int e = (int)(eh / Hh);
  int s, d;
  if (e < E) { s = src[e]; d = dst[e]; } else { s = d = e - E; }
  float alpha = EX[eh] / (DEN[d * Hh + h] + EPS_F);
  float val = Hm[(long long)s * Hh * C + (long long)h * C + c] * alpha;
  atomicAdd(&ACC[(long long)d * Hh * C + (long long)h * C + c], val);
}

// ---- layer finalizers ------------------------------------------------------

__global__ void k_fin_concat_elu(const float* __restrict__ ACC, const float* __restrict__ bias,
                                 _Float16* __restrict__ out16, int Nn, int HC) {
  long long idx = (long long)blockIdx.x * blockDim.x + threadIdx.x;
  if (idx >= (long long)Nn * HC) return;
  int j = (int)(idx % HC);
  float v = ACC[idx] + bias[j];
  v = v > 0.f ? v : expm1f(v);
  out16[idx] = (_Float16)v;
}

__global__ void k_fin_mean_elu(const float* __restrict__ ACC, const float* __restrict__ bias,
                               _Float16* __restrict__ out16, int Nn, int Hh, int C) {
  long long idx = (long long)blockIdx.x * blockDim.x + threadIdx.x;
  if (idx >= (long long)Nn * C) return;
  int n = (int)(idx / C), c = (int)(idx % C);
  float s = 0.f;
  for (int h = 0; h < Hh; ++h) s += ACC[(long long)n * Hh * C + (long long)h * C + c];
  float v = s / (float)Hh + bias[c];
  v = v > 0.f ? v : expm1f(v);
  out16[idx] = (_Float16)v;
}

__global__ void k_fin_out(const float* __restrict__ ACC, const float* __restrict__ bias,
                          float* __restrict__ out, int Nn, int C) {
  long long idx = (long long)blockIdx.x * blockDim.x + threadIdx.x;
  if (idx >= (long long)Nn * C) return;
  out[idx] = ACC[idx] + bias[(int)(idx % C)];
}

// ---------------------------------------------------------------------------

static inline unsigned gb(long long n, int t) { return (unsigned)((n + t - 1) / t); }

extern "C" void kernel_launch(void* const* d_in, const int* in_sizes, int n_in,
                              void* d_out, int out_size, void* d_ws, size_t ws_size,
                              hipStream_t stream) {
  constexpr int N = 50000, E = 800000, IN = 256, HID = 128, OUT = 128;
  constexpr int Ntot = E + N;

  const float* x    = (const float*)d_in[0];
  const int*   ei   = (const int*)d_in[1];
  const int*   src  = ei;
  const int*   dst  = ei + E;
  const float* W1   = (const float*)d_in[2];
  const float* b1   = (const float*)d_in[3];
  const float* W2   = (const float*)d_in[4];
  const float* b2   = (const float*)d_in[5];
  const float* g1W  = (const float*)d_in[6];
  const float* g1as = (const float*)d_in[7];
  const float* g1ad = (const float*)d_in[8];
  const float* g1b  = (const float*)d_in[9];
  const float* g2W  = (const float*)d_in[10];
  const float* g2as = (const float*)d_in[11];
  const float* g2ad = (const float*)d_in[12];
  const float* g2b  = (const float*)d_in[13];
  const float* g3W  = (const float*)d_in[14];
  const float* g3as = (const float*)d_in[15];
  const float* g3ad = (const float*)d_in[16];
  const float* g3b  = (const float*)d_in[17];
  float* out = (float*)d_out;
  (void)n_in; (void)in_sizes; (void)out_size; (void)ws_size;

  // ---- workspace layout ----
  char* ws = (char*)d_ws;
  size_t off = 0;
  auto walloc = [&](size_t bytes) -> void* {
    void* p = ws + off;
    off = (off + bytes + 255) & ~(size_t)255;
    return p;
  };
  _Float16* bufA = (_Float16*)walloc((size_t)N * 256 * sizeof(_Float16));
  _Float16* bufB = (_Float16*)walloc((size_t)N * 256 * sizeof(_Float16));
  _Float16* Wt   = (_Float16*)walloc((size_t)256 * 256 * sizeof(_Float16));
  float*    Hm   = (float*)walloc((size_t)N * 256 * sizeof(float));
  float*    ACC  = (float*)walloc((size_t)N * 256 * sizeof(float));
  float*    AS   = (float*)walloc((size_t)N * 4 * sizeof(float));
  float*    AD   = (float*)walloc((size_t)N * 4 * sizeof(float));
  unsigned* Mx   = (unsigned*)walloc((size_t)N * 4 * sizeof(unsigned));
  float*    DEN  = (float*)walloc((size_t)N * 4 * sizeof(float));
  float*    EX   = (float*)walloc((size_t)Ntot * 4 * sizeof(float));

  auto gemm = [&](const _Float16* A, const float* W, const float* bias, int K, int Nc,
                  int act, float* outF, _Float16* outH) {
    k_conv_w_t<<<gb((long long)K * Nc, 256), 256, 0, stream>>>(W, Wt, K, Nc);
    k_wmma_gemm<<<dim3((N + 127) / 128, Nc / 128), 256, 0, stream>>>(A, Wt, bias, outF,
                                                                     outH, N, K, Nc, act);
  };

  auto gat_edges = [&](int Hh, int C) {
    hipMemsetAsync(Mx, 0, (size_t)N * 4 * sizeof(unsigned), stream);
    hipMemsetAsync(DEN, 0, (size_t)N * 4 * sizeof(float), stream);
    hipMemsetAsync(ACC, 0, (size_t)N * 256 * sizeof(float), stream);
    long long nEH = (long long)Ntot * Hh;
    k_edge_max<<<gb(nEH, 256), 256, 0, stream>>>(src, dst, E, Ntot, AS, AD, Mx, Hh);
    k_edge_exp<<<gb(nEH, 256), 256, 0, stream>>>(src, dst, E, Ntot, AS, AD, Mx, DEN, EX, Hh);
    k_edge_scatter<<<gb(nEH * C, 256), 256, 0, stream>>>(src, dst, E, Ntot, Hm, EX, DEN,
                                                         ACC, Hh, C);
  };

  auto alpha = [&](const float* a_s, const float* a_d, int Hh, int C) {
    long long waves = (long long)N * Hh;
    k_alpha<<<gb(waves * 32, 256), 256, 0, stream>>>(Hm, a_s, a_d, AS, AD, N, Hh, C);
  };

  // ---- semantic MLP: h = relu(relu(x@W1+b1)@W2+b2), f16 activations ----
  k_f32_to_f16<<<gb((long long)N * IN, 256), 256, 0, stream>>>(x, bufA, (long long)N * IN);
  gemm(bufA, W1, b1, IN, HID, /*relu*/1, nullptr, bufB);
  gemm(bufB, W2, b2, HID, HID, /*relu*/1, nullptr, bufA);

  // ---- gat1: HID -> 4 heads x 64, concat, ELU ----
  gemm(bufA, g1W, nullptr, HID, 256, 0, Hm, nullptr);   // Hm = [N,4,64] f32
  alpha(g1as, g1ad, 4, 64);
  gat_edges(4, 64);
  k_fin_concat_elu<<<gb((long long)N * 256, 256), 256, 0, stream>>>(ACC, g1b, bufB, N, 256);

  // ---- gat2: 256 -> 2 heads x 128, mean, ELU ----
  gemm(bufB, g2W, nullptr, 256, 256, 0, Hm, nullptr);   // Hm = [N,2,128] f32
  alpha(g2as, g2ad, 2, 128);
  gat_edges(2, 128);
  k_fin_mean_elu<<<gb((long long)N * 128, 256), 256, 0, stream>>>(ACC, g2b, bufA, N, 2, 128);

  // ---- gat3: 128 -> 1 head x 128, mean, no activation ----
  gemm(bufA, g3W, nullptr, 128, 128, 0, Hm, nullptr);   // Hm = [N,1,128] f32
  alpha(g3as, g3ad, 1, 128);
  gat_edges(1, 128);
  k_fin_out<<<gb((long long)N * OUT, 256), 256, 0, stream>>>(ACC, g3b, out, N, OUT);
}
